// Encoder_8718783611479
// MI455X (gfx1250) — compile-verified, matrix operand order checked by reading, not run on hardware
//
#include <hip/hip_runtime.h>

typedef __attribute__((ext_vector_type(16))) _Float16 v16h;
typedef __attribute__((ext_vector_type(8)))  float    v8f;

// Problem dims (from reference)
#define B_SZ   2048
#define T_SZ   240
#define K_DIM  180          // D_IN
#define H_SZ   90           // hidden per gate
// Tiling
#define KCHUNKS 6           // K padded 180 -> 192 = 6 x 32
#define NGATE   3           // i, g, o  (f gate multiplies c_prev == 0 -> dead)
#define NPAD    96          // per-gate columns padded 90 -> 96 = 6 x 16
#define NTILES  (NGATE * 6) // 18 accumulator tiles of 16x16
#define NFRAGS  (NTILES * KCHUNKS)        // 108 B-fragments
#define MTILES_TOTAL (B_SZ * T_SZ / 16)   // 30720 (240/16 = 15, tiles never cross b)
#define TPB     256
#define WAVES   8
#define MT_PER_WAVE 4
#define NBLOCKS (MTILES_TOTAL / (WAVES * MT_PER_WAVE)) // 960

#define FRAG_HALFS 16                               // halves per lane per fragment (32B)
#define WFRAG_HALFS (NFRAGS * 32 * FRAG_HALFS)      // 55296 halves = 108 KB
#define SMEM_BYTES  (WFRAG_HALFS * 2 + NGATE * NPAD * 4)  // + bias table

__device__ __forceinline__ float fast_sigmoid(float x) {
    return 1.0f / (1.0f + __expf(-x));     // v_exp_f32 path
}
__device__ __forceinline__ float fast_tanh(float x) {
    float e = __expf(2.0f * x);
    return (e - 1.0f) / (e + 1.0f);
}
__device__ __forceinline__ int gate_row_base(int gate) {
    // torch gate order i,f,g,o -> row bases 0,90,180,270 ; we use i,g,o
    return gate == 0 ? 0 : (gate == 1 ? 180 : 270);
}

extern "C" __global__ __launch_bounds__(TPB)
void encoder_lstm_fused(const float* __restrict__ x,     // (B,3,60,240) == (B, K=180, T=240)
                        const float* __restrict__ W_ih,  // (360, 180) row-major
                        const float* __restrict__ b_ih,  // (360,)
                        const float* __restrict__ b_hh,  // (360,)
                        float* __restrict__ out)         // (B, T, 90)
{
    extern __shared__ _Float16 smem[];
    _Float16* wfrag   = smem;                           // B fragments, WMMA lane order
    float*    biasLds = (float*)(smem + WFRAG_HALFS);   // combined b_ih + b_hh, padded

    const int tid   = threadIdx.x;
    const int lane  = tid & 31;
    const int wave  = tid >> 5;
    const int lhalf = lane >> 4;   // lane 0-15 vs 16-31
    const int nl    = lane & 15;

    // ---------------- Stage W into LDS in exact B-fragment order (f16) -------------
    // Flat layout: ((nt*KCHUNKS + kc)*32 + lane)*16 + (2*j + e)
    // B(k,n) assumed mirror of 16-bit A layout with N in place of M:
    //   lanes 0-15:  n = nl, K = {0..7, 16..23} ; lanes 16-31: same n, K ranges +8/+24
    // One 32B fragment slice per thread per iteration: decode amortized over 16 halves,
    // global reads are two contiguous 8-float runs of a W row, LDS store is one v16h.
    for (int fi = wave; fi < NFRAGS; fi += WAVES) {
        int nt = fi / KCHUNKS, kc = fi % KCHUNKS;
        int gate = nt / 6, ct = nt % 6;
        int n = ct * 16 + nl;
        bool nok = (n < H_SZ);
        const float* wrow = W_ih + (gate_row_base(gate) + (nok ? n : 0)) * K_DIM;
        v16h frag;
        #pragma unroll
        for (int h = 0; h < 16; ++h) {
            int j = h >> 1, e = h & 1;
            int kb = (j < 4 ? 2 * j : 16 + 2 * (j - 4));
            int k = kc * 32 + kb + 8 * lhalf + e;
            float v = (nok && k < K_DIM) ? wrow[k] : 0.0f;
            frag[h] = (_Float16)v;
        }
        *(v16h*)(wfrag + fi * (32 * FRAG_HALFS) + lane * FRAG_HALFS) = frag;
    }
    for (int idx = tid; idx < NGATE * NPAD; idx += TPB) {
        int gate = idx / NPAD, n = idx % NPAD;
        float v = 0.0f;
        if (n < H_SZ) {
            int r = gate_row_base(gate) + n;
            v = b_ih[r] + b_hh[r];
        }
        biasLds[idx] = v;
    }

    #pragma clang loop unroll(disable)
    for (int jt = 0; jt < MT_PER_WAVE; ++jt) {
        // Barrier every iteration: (a) first one closes the staging phase,
        // (b) its LDS-fence semantics stop LICM from hoisting the 108 B-fragment
        //     ds_loads out of this loop (which would spill 800+ VGPRs to scratch).
        __syncthreads();

        const int mtile = (blockIdx.x * WAVES + wave) * MT_PER_WAVE + jt;
        const int b  = mtile / 15;
        const int t0 = (mtile % 15) * 16;
        // Lane's A-row base: x[b][k][t0 + nl]; loads below use immediate offsets k*240*4
        const float* xb = x + (size_t)b * (K_DIM * T_SZ) + t0 + nl;

        // Prefetch next tile's A stream while we compute this one
        if (jt + 1 < MT_PER_WAVE) {
            int mt2 = mtile + 1;
            const float* xn = x + (size_t)(mt2 / 15) * (K_DIM * T_SZ) + (mt2 % 15) * 16 + nl;
            __builtin_prefetch(xn, 0, 0);   // global_prefetch_b8
        }

        v8f acc[NTILES] = {};   // 18 tiles x 8 VGPRs f32

        #pragma unroll
        for (int kc = 0; kc < KCHUNKS; ++kc) {
            // ---- A fragment: 16 half elements per lane, coalesced f32 loads + cvt ----
            v16h a;
            #pragma unroll
            for (int hi = 0; hi < 16; ++hi) {
                int j = hi >> 1, e = hi & 1;
                int kbase = (j < 4 ? 2 * j : 16 + 2 * (j - 4)) + 8 * lhalf;
                int k = kc * 32 + kbase + e;                 // only kc==5 keeps the mask
                float v = (k < K_DIM) ? xb[k * T_SZ] : 0.0f; // contiguous 64B per k
                a[hi] = (_Float16)v;
            }
            // ---- 18 WMMAs reuse this A fragment; B fragments stream from LDS ----
            const _Float16* wb = wfrag + ((kc * 32 + lane) << 4);
            #pragma unroll
            for (int nt = 0; nt < NTILES; ++nt) {
                v16h bf = *(const v16h*)(wb + nt * (KCHUNKS * 32 * FRAG_HALFS));
                acc[nt] = __builtin_amdgcn_wmma_f32_16x16x32_f16(
                    false, a, false, bf, (short)0, acc[nt], false, false);
            }
        }

        // ---------------- Fused LSTM-cell epilogue + store ----------------
        // C/D tile layout: VGPR r, lanes 0-15 -> (M=r, N=nl), lanes 16-31 -> (M=r+8, N=nl)
        const size_t mbase = (size_t)mtile * 16;
        #pragma unroll
        for (int ct = 0; ct < 6; ++ct) {
            int n = ct * 16 + nl;
            if (n < H_SZ) {
                float bi = biasLds[n];
                float bg = biasLds[NPAD + n];
                float bo = biasLds[2 * NPAD + n];
                #pragma unroll
                for (int r = 0; r < 8; ++r) {
                    float iv = acc[ct][r]      + bi;
                    float gv = acc[6 + ct][r]  + bg;
                    float ov = acc[12 + ct][r] + bo;
                    float c  = fast_sigmoid(iv) * fast_tanh(gv);  // f*c_prev == 0
                    float hh = fast_sigmoid(ov) * fast_tanh(c);
                    float y  = fast_sigmoid(hh);
                    size_t m = mbase + (size_t)(r + 8 * lhalf);
                    out[m * H_SZ + n] = y;
                }
            }
        }
    }
}

extern "C" void kernel_launch(void* const* d_in, const int* in_sizes, int n_in,
                              void* d_out, int out_size, void* d_ws, size_t ws_size,
                              hipStream_t stream) {
    (void)in_sizes; (void)n_in; (void)out_size; (void)d_ws; (void)ws_size;
    const float* x    = (const float*)d_in[0];
    const float* W_ih = (const float*)d_in[1];
    // d_in[2] = W_hh: provably unused (h0 = c0 = 0 at every step)
    const float* b_ih = (const float*)d_in[3];
    const float* b_hh = (const float*)d_in[4];
    float* out = (float*)d_out;

    // 108 KB dynamic LDS (> default cap) — raise the limit every call
    // (deterministic host-side call; not a stream op, safe under graph capture).
    hipFuncSetAttribute((const void*)encoder_lstm_fused,
                        hipFuncAttributeMaxDynamicSharedMemorySize, SMEM_BYTES);

    encoder_lstm_fused<<<NBLOCKS, TPB, SMEM_BYTES, stream>>>(x, W_ih, b_ih, b_hh, out);
}
